// GatherModel_63101659513270
// MI455X (gfx1250) — matrix operand order repaired and checked by reading, not compiled, not run on hardware
//
#include <hip/hip_runtime.h>
#include <hip/hip_bf16.h>

typedef __attribute__((ext_vector_type(16))) _Float16 v16h;
typedef __attribute__((ext_vector_type(8)))  float    v8f;
typedef __attribute__((ext_vector_type(4)))  unsigned v4u;

#define NN 100000            // nodes
#define NE 1600000           // edges without self loops
#define ET (NE + NN)         // edges incl. self loops
#define NH 4                 // heads
#define NC 42                // channels / head
#define HC 168               // NH*NC
#define HP 176               // padded h row stride (11 * 16)
#define MT (NN / 16)         // 6250 row tiles (exact)
#define NG 64                // graphs

static __device__ __forceinline__ unsigned f2ord(float f) {
  unsigned u = __float_as_uint(f);
  return (u & 0x80000000u) ? ~u : (u | 0x80000000u);
}
static __device__ __forceinline__ float ord2f(unsigned u) {
  return __uint_as_float((u & 0x80000000u) ? (u & 0x7fffffffu) : ~u);
}
static __device__ __forceinline__ float lrelu(float v) { return v > 0.f ? v : 0.2f * v; }
static __device__ __forceinline__ float eluf(float v)  { return v > 0.f ? v : expm1f(v); }

// ---------------------------------------------------------------- utilities
__global__ void zero_f32(float* __restrict__ p, int n) {
  int i = blockIdx.x * blockDim.x + threadIdx.x;
  if (i < n) p[i] = 0.f;
}

// x [NN,42] f32 -> XF16 [NN,64] f16 zero-padded
__global__ void prep_x(const float* __restrict__ x, _Float16* __restrict__ xf) {
  int i = blockIdx.x * blockDim.x + threadIdx.x;
  if (i >= NN * 64) return;
  int n = i >> 6, c = i & 63;
  xf[i] = (_Float16)(c < NC ? x[n * NC + c] : 0.f);
}

// W [fin,fout] f32 -> Wt [foutPad, kPad] f16 (transposed, zero-padded)
__global__ void prep_w(const float* __restrict__ W, _Float16* __restrict__ Wt,
                       int fin, int fout, int kPad, int foutPad) {
  int i = blockIdx.x * blockDim.x + threadIdx.x;
  if (i >= foutPad * kPad) return;
  int col = i / kPad, k = i % kPad;
  Wt[i] = (_Float16)((col < fout && k < fin) ? W[k * fout + col] : 0.f);
}

// ---------------------------------------------------------------- WMMA GEMM
// One wave per 16-row panel. The full-K A fragment (KSTEPS*8 VGPRs) is loaded
// once and reused across all NT column tiles -> steady state is 2 b128 loads
// (weights only, L2-resident) per v_wmma_f32_16x16x32_f16.
template <int KSTEPS, int NT>
__global__ __launch_bounds__(32)
void gemm_wmma(const _Float16* __restrict__ A, const _Float16* __restrict__ Bt,
               float* __restrict__ C) {
  const int lane = threadIdx.x;
  const int hi   = lane >> 4;          // half-wave select
  const int lo   = lane & 15;
  const int kPad = KSTEPS * 32;
  const _Float16* arow = A + (size_t)(blockIdx.x * 16 + lo) * kPad;

  // A 16x32 f16 fragment per kstep: lane(lo)=M row; regs 0..3 hold
  // K=hi*8+0..7, regs 4..7 hold K=16+hi*8+0..7
  union U { v16h h; v4u u[2]; };
  U a[KSTEPS];
#pragma unroll
  for (int ks = 0; ks < KSTEPS; ++ks) {
    a[ks].u[0] = *(const v4u*)(arow + ks * 32 + hi * 8);
    a[ks].u[1] = *(const v4u*)(arow + ks * 32 + 16 + hi * 8);
  }

#pragma unroll
  for (int t = 0; t < NT; ++t) {
    const _Float16* brow = Bt + (size_t)(t * 16 + lo) * kPad;
    v8f acc = {};
#pragma unroll
    for (int ks = 0; ks < KSTEPS; ++ks) {
      // B 32x16 f16 fragment: lane(lo)=N col; regs 0..3 K=hi*8+0..7, regs 4..7 K=16+hi*8+0..7
      U b;
      const _Float16* bp = brow + ks * 32 + hi * 8;
      b.u[0] = *(const v4u*)(bp);
      b.u[1] = *(const v4u*)(bp + 16);
      acc = __builtin_amdgcn_wmma_f32_16x16x32_f16(false, a[ks].h, false, b.h,
                                                   (short)0, acc, false, false);
    }
    // D layout: reg r, lane -> row = r + 8*hi, col = lo
    const int col = t * 16 + lo;
    const size_t rbase = (size_t)(blockIdx.x * 16 + hi * 8);
#pragma unroll
    for (int r = 0; r < 8; ++r)
      C[(rbase + r) * HP + col] = acc[r];
  }
}

// ---------------------------------------------------------------- attention
__global__ void alpha_kernel(const float* __restrict__ Hm, const float* __restrict__ as,
                             const float* __restrict__ ad, float* __restrict__ As,
                             float* __restrict__ Ad) {
  int i = blockIdx.x * blockDim.x + threadIdx.x;   // n*NH + h
  if (i >= NN * NH) return;
  int n = i >> 2, h = i & 3;
  const float* hp = Hm + (size_t)n * HP + h * NC;
  float s = 0.f, d = 0.f;
  for (int c = 0; c < NC; ++c) { float v = hp[c]; s += v * as[h * NC + c]; d += v * ad[h * NC + c]; }
  As[i] = s; Ad[i] = d;
}

__global__ void edge_max_kernel(const int* __restrict__ src, const int* __restrict__ dst,
                                const float* __restrict__ As, const float* __restrict__ Ad,
                                unsigned* __restrict__ Mx) {
  int i = blockIdx.x * blockDim.x + threadIdx.x;
  if (i >= ET * NH) return;
  int e = i >> 2, h = i & 3;
  int s = (e < NE) ? src[e] : (e - NE);
  int d = (e < NE) ? dst[e] : (e - NE);
  float v = lrelu(As[s * NH + h] + Ad[d * NH + h]);
  atomicMax(&Mx[d * NH + h], f2ord(v));
}

__global__ void edge_sum_kernel(const int* __restrict__ src, const int* __restrict__ dst,
                                const float* __restrict__ As, const float* __restrict__ Ad,
                                const unsigned* __restrict__ Mx, float* __restrict__ Den) {
  int i = blockIdx.x * blockDim.x + threadIdx.x;
  if (i >= ET * NH) return;
  int e = i >> 2, h = i & 3;
  int s = (e < NE) ? src[e] : (e - NE);
  int d = (e < NE) ? dst[e] : (e - NE);
  float v = lrelu(As[s * NH + h] + Ad[d * NH + h]);
  atomicAdd(&Den[d * NH + h], __expf(v - ord2f(Mx[d * NH + h])));
}

// one wave per edge: Acc[dst] += h[src] * alpha  (accumulator lives in L2)
__global__ __launch_bounds__(256)
void edge_aggr_kernel(const int* __restrict__ src, const int* __restrict__ dst,
                      const float* __restrict__ As, const float* __restrict__ Ad,
                      const unsigned* __restrict__ Mx, const float* __restrict__ Den,
                      const float* __restrict__ Hm, float* __restrict__ Acc) {
  int wave = (blockIdx.x * blockDim.x + threadIdx.x) >> 5;
  int lane = threadIdx.x & 31;
  if (wave >= ET) return;
  int s = (wave < NE) ? src[wave] : (wave - NE);
  int d = (wave < NE) ? dst[wave] : (wave - NE);
  float alpha[NH];
#pragma unroll
  for (int h = 0; h < NH; ++h) {
    float v = lrelu(As[s * NH + h] + Ad[d * NH + h]);
    alpha[h] = __expf(v - ord2f(Mx[d * NH + h])) / Den[d * NH + h];
  }
  const float* hrow = Hm + (size_t)s * HP;
  float* arow = Acc + (size_t)d * HP;
  for (int c = lane; c < HC; c += 32)
    atomicAdd(&arow[c], hrow[c] * alpha[c / NC]);
}

// ---------------------------------------------------------------- epilogues
// elu(acc + bias) -> f16 activations with zero-padded stride for next GEMM
__global__ void fin_f16(const float* __restrict__ Acc, const float* __restrict__ bias,
                        _Float16* __restrict__ Xo, int fout, int padNext) {
  int i = blockIdx.x * blockDim.x + threadIdx.x;
  if (i >= NN * padNext) return;
  int n = i / padNext, c = i % padNext;
  float v = 0.f;
  if (c < fout) v = eluf(Acc[(size_t)n * HP + c] + bias[c]);
  Xo[i] = (_Float16)v;
}

__global__ void count_kernel(const int* __restrict__ batch, float* __restrict__ Cnt) {
  int n = blockIdx.x * blockDim.x + threadIdx.x;
  if (n < NN) atomicAdd(&Cnt[batch[n]], 1.f);
}

__global__ void fin_pool(const float* __restrict__ Acc, const float* __restrict__ bias,
                         const int* __restrict__ batch, float* __restrict__ Pool) {
  int i = blockIdx.x * blockDim.x + threadIdx.x;
  if (i >= NN * HC) return;
  int n = i / HC, c = i % HC;
  float v = eluf(Acc[(size_t)n * HP + c] + bias[c]);
  atomicAdd(&Pool[batch[n] * HC + c], v);
}

__global__ void pool_div(const float* __restrict__ Pool, const float* __restrict__ Cnt,
                         float* __restrict__ out) {
  int i = blockIdx.x * blockDim.x + threadIdx.x;
  if (i >= NG * HC) return;
  out[i] = Pool[i] / fmaxf(Cnt[i / HC], 1.f);
}

// ---------------------------------------------------------------- launcher
#define CDIV(a, b) (((a) + (b) - 1) / (b))

extern "C" void kernel_launch(void* const* d_in, const int* in_sizes, int n_in,
                              void* d_out, int out_size, void* d_ws, size_t ws_size,
                              hipStream_t stream) {
  (void)in_sizes; (void)n_in; (void)out_size; (void)ws_size;
  const float* x     = (const float*)d_in[0];
  const int*   ei    = (const int*)d_in[1];
  const int*   src   = ei;
  const int*   dst   = ei + NE;
  const int*   batch = (const int*)d_in[2];
  const float* l0w   = (const float*)d_in[3];
  const float* l0b   = (const float*)d_in[4];
  const float* Wl[3]  = {(const float*)d_in[5],  (const float*)d_in[9],  (const float*)d_in[13]};
  const float* ASl[3] = {(const float*)d_in[6],  (const float*)d_in[10], (const float*)d_in[14]};
  const float* ADl[3] = {(const float*)d_in[7],  (const float*)d_in[11], (const float*)d_in[15]};
  const float* Bl[3]  = {(const float*)d_in[8],  (const float*)d_in[12], (const float*)d_in[16]};
  float* out = (float*)d_out;

  size_t off = 0;
  auto carve = [&](size_t bytes) -> void* {
    void* p = (void*)((char*)d_ws + off);
    off += (bytes + 255) & ~(size_t)255;
    return p;
  };
  _Float16* XF   = (_Float16*)carve((size_t)NN * 192 * 2);   // activations f16 (padded)
  float*    Hf   = (float*)   carve((size_t)NN * HP * 4);    // h = x@W
  float*    Ac   = (float*)   carve((size_t)NN * HP * 4);    // aggregation accumulator
  float*    Asrc = (float*)   carve((size_t)NN * NH * 4);
  float*    Adst = (float*)   carve((size_t)NN * NH * 4);
  unsigned* Mx   = (unsigned*)carve((size_t)NN * NH * 4);
  float*    Den  = (float*)   carve((size_t)NN * NH * 4);
  _Float16* Wt   = (_Float16*)carve((size_t)HP * 192 * 2);
  float*    Pool = (float*)   carve((size_t)NG * HC * 4);
  float*    Cnt  = (float*)   carve((size_t)NG * 4);

  const int T = 256;

  // ---- lin0: x = elu(x @ lin0_w + b)     [Fin 42->pad64, Fout 42->pad48]
  prep_x<<<CDIV(NN * 64, T), T, 0, stream>>>(x, XF);
  prep_w<<<CDIV(48 * 64, T), T, 0, stream>>>(l0w, Wt, 42, 42, 64, 48);
  gemm_wmma<2, 3><<<MT, 32, 0, stream>>>(XF, Wt, Hf);
  fin_f16<<<CDIV(NN * 64, T), T, 0, stream>>>(Hf, l0b, XF, 42, 64);

  // ---- 3 GAT layers
  for (int l = 0; l < 3; ++l) {
    const int fin  = (l == 0) ? 42 : 168;
    const int kPad = (l == 0) ? 64 : 192;
    prep_w<<<CDIV(HP * kPad, T), T, 0, stream>>>(Wl[l], Wt, fin, HC, kPad, HP);
    if (l == 0) gemm_wmma<2, 11><<<MT, 32, 0, stream>>>(XF, Wt, Hf);
    else        gemm_wmma<6, 11><<<MT, 32, 0, stream>>>(XF, Wt, Hf);
    alpha_kernel<<<CDIV(NN * NH, T), T, 0, stream>>>(Hf, ASl[l], ADl[l], Asrc, Adst);
    zero_f32<<<CDIV(NN * NH, T), T, 0, stream>>>((float*)Mx, NN * NH);
    zero_f32<<<CDIV(NN * NH, T), T, 0, stream>>>(Den, NN * NH);
    zero_f32<<<CDIV(NN * HP, T), T, 0, stream>>>(Ac, NN * HP);
    edge_max_kernel<<<CDIV(ET * NH, T), T, 0, stream>>>(src, dst, Asrc, Adst, Mx);
    edge_sum_kernel<<<CDIV(ET * NH, T), T, 0, stream>>>(src, dst, Asrc, Adst, Mx, Den);
    edge_aggr_kernel<<<CDIV(ET * 32, T), T, 0, stream>>>(src, dst, Asrc, Adst, Mx, Den, Hf, Ac);
    if (l < 2) {
      fin_f16<<<CDIV(NN * 192, T), T, 0, stream>>>(Ac, Bl[l], XF, HC, 192);
    } else {
      zero_f32<<<CDIV(NG * HC, T), T, 0, stream>>>(Pool, NG * HC);
      zero_f32<<<1, T, 0, stream>>>(Cnt, NG);
      count_kernel<<<CDIV(NN, T), T, 0, stream>>>(batch, Cnt);
      fin_pool<<<CDIV(NN * HC, T), T, 0, stream>>>(Ac, Bl[2], batch, Pool);
      pool_div<<<CDIV(NG * HC, T), T, 0, stream>>>(Pool, Cnt, out);
    }
  }
}
